// FullAttention_11132555231875
// MI455X (gfx1250) — matrix operand order repaired and checked by reading, not compiled, not run on hardware
//
#include <hip/hip_runtime.h>
#include <hip/hip_bf16.h>

typedef __attribute__((ext_vector_type(16))) _Float16 v16h;
typedef __attribute__((ext_vector_type(8)))  _Float16 v8h;
typedef __attribute__((ext_vector_type(8)))  float    v8f;
typedef __attribute__((ext_vector_type(4)))  float    v4f;

// B=4, L=2048, H=16, E=64 (fp32 in/out).
// Block = 8 waves = 128 consecutive query rows of one (b,h).
// Cooperative: K tile (32x64 f16) + V^T tile (64x32 f16) staged in LDS once
// per key block, shared by all 8 waves. Per-wave: WMMA QK^T, online softmax,
// P transpose through private LDS, WMMA P*V.

__device__ __forceinline__ v16h ld16(const _Float16* p) {
    v8h a = *(const v8h*)p;
    v8h b = *(const v8h*)(p + 8);
    v16h r;
    #pragma unroll
    for (int i = 0; i < 8; ++i) { r[i] = a[i]; r[i + 8] = b[i]; }
    return r;
}

__global__ __launch_bounds__(256)
void fa_causal_wmma_kernel(const float* __restrict__ Qg,
                           const float* __restrict__ Kg,
                           const float* __restrict__ Vg,
                           float* __restrict__ Og) {
    constexpr int L = 2048, H = 16, E = 64;
    constexpr int ROWSTRIDE = H * E;                    // floats per l step
    constexpr float CS = 0.125f * 1.4426950408889634f;  // (1/sqrt(E))*log2(e)
    constexpr int KS = 72;   // K-tile LDS row stride (halves), 144B rows
    constexpr int VS = 40;   // V^T-tile LDS row stride (halves), 80B rows
    constexpr int PS = 40;   // P-tile LDS row stride (halves)

    const int t    = threadIdx.x;
    const int lane = t & 31;
    const int wave = t >> 5;
    const int hi   = lane >> 4;
    const int ln   = lane & 15;

    // blockIdx -> (b, h, query group of 128 rows)
    const int qg    = blockIdx.x & 15;     // 16 groups of 128 rows
    const int bh    = blockIdx.x >> 4;     // 0..63
    const int h     = bh & 15;
    const int b     = bh >> 4;
    const int qgB   = qg << 7;             // group base row
    const int qbase = qgB + (wave << 4);   // this wave's 16 query rows

    const size_t bhOff = ((size_t)b * L) * ROWSTRIDE + (size_t)h * E;

    __shared__ _Float16 kLds[32 * KS];           // K rows (s-major, e contiguous)
    __shared__ _Float16 vLds[64 * VS];           // V transposed (d-major, s contiguous)
    __shared__ _Float16 pLds[8 * 16 * PS];       // per-wave P transpose buffers
    _Float16* pw = &pLds[wave * 16 * PS];

    // ---- Q A-fragments (loop invariant)
    const float* qrow = Qg + bhOff + (size_t)(qbase + ln) * ROWSTRIDE;
    v16h aq0, aq1;
    #pragma unroll
    for (int i = 0; i < 8; ++i) {
        aq0[i]     = (_Float16)qrow[hi * 16 + i];
        aq0[i + 8] = (_Float16)qrow[hi * 16 + 8 + i];
        aq1[i]     = (_Float16)qrow[32 + hi * 16 + i];
        aq1[i + 8] = (_Float16)qrow[32 + hi * 16 + 8 + i];
    }

    v8f o[4];
    float mrow[8], lrow[8];
    #pragma unroll
    for (int nt = 0; nt < 4; ++nt)
        #pragma unroll
        for (int v = 0; v < 8; ++v) o[nt][v] = 0.0f;
    #pragma unroll
    for (int v = 0; v < 8; ++v) { mrow[v] = -__builtin_inff(); lrow[v] = 0.0f; }

    const int nbShared = (qgB + 128) >> 5;   // key blocks for whole group
    const int sr = t >> 3;                   // staging: row 0..31
    const int sc = (t & 7) << 3;             // staging: col 0..56 step 8

    for (int blk = 0; blk < nbShared; ++blk) {
        const int sbase = blk << 5;

        __syncthreads();   // previous block's LDS reads done (WAR)

        // ---- cooperative staging: K tile -> kLds (f16), V tile -> vLds (f16^T)
        {
            const float* ksrc = Kg + bhOff + (size_t)(sbase + sr) * ROWSTRIDE + sc;
            v4f k0 = *(const v4f*)ksrc;
            v4f k1 = *(const v4f*)(ksrc + 4);
            v8h kp;
            #pragma unroll
            for (int i = 0; i < 4; ++i) { kp[i] = (_Float16)k0[i]; kp[i + 4] = (_Float16)k1[i]; }
            *(v8h*)(&kLds[sr * KS + sc]) = kp;

            const float* vsrc = Vg + bhOff + (size_t)(sbase + sr) * ROWSTRIDE + sc;
            v4f w0 = *(const v4f*)vsrc;
            v4f w1 = *(const v4f*)(vsrc + 4);
            #pragma unroll
            for (int i = 0; i < 4; ++i) {
                vLds[(sc + i) * VS + sr]     = (_Float16)w0[i];
                vLds[(sc + 4 + i) * VS + sr] = (_Float16)w1[i];
            }
        }

        __syncthreads();   // staging visible (RAW)

        if (sbase >= qbase + 16) continue;   // wave-uniform: fully masked block

        // ---- S = Q * K^T  (two 16x16 f32 tiles)
        v8f s[2];
        #pragma unroll
        for (int jt = 0; jt < 2; ++jt)
            #pragma unroll
            for (int v = 0; v < 8; ++v) s[jt][v] = 0.0f;

        #pragma unroll
        for (int jt = 0; jt < 2; ++jt) {
            const _Float16* kr = &kLds[(jt * 16 + ln) * KS + hi * 16];
            v16h bk0 = ld16(kr);
            v16h bk1 = ld16(kr + 32);
            s[jt] = __builtin_amdgcn_wmma_f32_16x16x32_f16(false, aq0, false, bk0,
                                                           (short)0, s[jt], false, false);
            s[jt] = __builtin_amdgcn_wmma_f32_16x16x32_f16(false, aq1, false, bk1,
                                                           (short)0, s[jt], false, false);
        }

        // ---- causal mask on diagonal-crossing blocks
        if (sbase + 31 > qbase) {
            #pragma unroll
            for (int jt = 0; jt < 2; ++jt) {
                const int scol = sbase + jt * 16 + ln;
                #pragma unroll
                for (int v = 0; v < 8; ++v) {
                    const int qi = qbase + v + 8 * hi;
                    if (scol > qi) s[jt][v] = -__builtin_inff();
                }
            }
        }

        // ---- online softmax
        float bm[8];
        #pragma unroll
        for (int v = 0; v < 8; ++v) bm[v] = fmaxf(s[0][v], s[1][v]);
        #pragma unroll
        for (int off = 8; off >= 1; off >>= 1)
            #pragma unroll
            for (int v = 0; v < 8; ++v)
                bm[v] = fmaxf(bm[v], __shfl_xor(bm[v], off, 32));

        float alpha[8];
        #pragma unroll
        for (int v = 0; v < 8; ++v) {
            const float nm = fmaxf(mrow[v], bm[v]);
            alpha[v] = exp2f((mrow[v] - nm) * CS);
            mrow[v]  = nm;
        }

        float rs[8];
        #pragma unroll
        for (int v = 0; v < 8; ++v) rs[v] = 0.0f;
        #pragma unroll
        for (int jt = 0; jt < 2; ++jt)
            #pragma unroll
            for (int v = 0; v < 8; ++v) {
                const float p = exp2f((s[jt][v] - mrow[v]) * CS);
                rs[v] += p;
                pw[(v + 8 * hi) * PS + jt * 16 + ln] = (_Float16)p;
            }
        #pragma unroll
        for (int off = 8; off >= 1; off >>= 1)
            #pragma unroll
            for (int v = 0; v < 8; ++v)
                rs[v] += __shfl_xor(rs[v], off, 32);
        #pragma unroll
        for (int v = 0; v < 8; ++v) lrow[v] = lrow[v] * alpha[v] + rs[v];

        #pragma unroll
        for (int nt = 0; nt < 4; ++nt)
            #pragma unroll
            for (int v = 0; v < 8; ++v) o[nt][v] *= alpha[v];

        // ---- P back in A-layout (private LDS region, DS in-order per wave)
        v16h ap;
        {
            const int rb = ln * PS;
            #pragma unroll
            for (int i = 0; i < 8; ++i) {
                ap[i]     = pw[rb + hi * 16 + i];
                ap[i + 8] = pw[rb + hi * 16 + 8 + i];
            }
        }

        // ---- O += P * V  (V^T fragments are contiguous LDS reads)
        #pragma unroll
        for (int nt = 0; nt < 4; ++nt) {
            v16h bv = ld16(&vLds[(nt * 16 + ln) * VS + hi * 16]);
            o[nt] = __builtin_amdgcn_wmma_f32_16x16x32_f16(false, ap, false, bv,
                                                           (short)0, o[nt], false, false);
        }
    }

    // ---- normalize and store fp32 [B,L,H,E]
    #pragma unroll
    for (int v = 0; v < 8; ++v) {
        const float inv = 1.0f / lrow[v];
        float* orow = Og + bhOff + (size_t)(qbase + v + 8 * hi) * ROWSTRIDE + ln;
        #pragma unroll
        for (int nt = 0; nt < 4; ++nt)
            orow[nt * 16] = o[nt][v] * inv;
    }
}

extern "C" void kernel_launch(void* const* d_in, const int* in_sizes, int n_in,
                              void* d_out, int out_size, void* d_ws, size_t ws_size,
                              hipStream_t stream) {
    const float* Q = (const float*)d_in[0];
    const float* K = (const float*)d_in[1];
    const float* V = (const float*)d_in[2];
    float* O = (float*)d_out;
    // blocks = B*H * (L/128) = 64 * 16 = 1024; 256 threads (8 waves)
    dim3 grid(1024), block(256);
    fa_causal_wmma_kernel<<<grid, block, 0, stream>>>(Q, K, V, O);
}
